// RSCM_Loss_68195490726264
// MI455X (gfx1250) — matrix operand order repaired and checked by reading, not compiled, not run on hardware
//
#include <hip/hip_runtime.h>
#include <hip/hip_bf16.h>
#include <math.h>

// ---------------------------------------------------------------------------
// Problem constants (reference: N=1024, C=200, A=256; P=0.8, Q=2.0, EPS=0.01)
// ---------------------------------------------------------------------------
#define NN   1024
#define CC   200
#define AA   256
#define CPAD 208          // C padded to 13 tiles of 16
#define CVST 264          // CV^T row stride in bf16 elems (16B-aligned rows)

typedef __attribute__((ext_vector_type(16))) __bf16 v16bf;
typedef __attribute__((ext_vector_type(8)))  __bf16 v8bf;
typedef __attribute__((ext_vector_type(4)))  __bf16 v4bf;
typedef __attribute__((ext_vector_type(8)))  float  v8f;

// LDS carve-up for the per-class kernel (total 244,544 B <= 320 KB WGP LDS)
constexpr unsigned SD_OFF  = 0;                               // bf16 D [CPAD][AA]   = 106,496 B
constexpr unsigned CVT_OFF = SD_OFF + CPAD * AA * 2;          // bf16 CV^T [AA][CVST]= 135,168 B
constexpr unsigned SIG_OFF = CVT_OFF + AA * CVST * 2;         // f32 sigma accum [CPAD]
constexpr unsigned SWK_OFF = SIG_OFF + CPAD * 4;              // f32 W[k] row [AA]
constexpr unsigned SDM_OFF = SWK_OFF + AA * 4;                // f32 dmean [AA]
constexpr unsigned LDS_BYTES = SDM_OFF + AA * 4;              // = 244,544

// ---------------------------------------------------------------------------
// Kernel 1: per-class precompute.  add[k,c] = 0.5*L*quad[k,c] + L*shift[k,c]
//   quad[k,c] = sum_{a,b} D[c,a] CV[k,a,b] D[c,b],  D = W - W[k]
// GEMM P = D @ CV via v_wmma_f32_16x16x32_bf16 (f32 accumulate).
//  - A fragments cached in 64 VGPRs, loaded once per ct row
//  - per bt tile: sched_group_barrier pins [24 DS reads][8 WMMA] so the
//    compiler emits staged s_wait_dscnt counts (loads overlap the WMMA chain)
//  - D bits for the P.*D rowsum loaded first -> wait-free VALU tail
//  - fold once per ct with 8 ds_add_f32 atomics
// ---------------------------------------------------------------------------
__global__ __launch_bounds__(256) void rscm_perclass_kernel(
    const float* __restrict__ W,        // [CC,AA]
    const float* __restrict__ mean_s,   // [CC,AA]
    const float* __restrict__ mean_t,   // [CC,AA]
    const float* __restrict__ cov,      // [CC,AA,AA]
    const float* __restrict__ LambdaP,  // [1]
    float* __restrict__ addOut)         // [CC,CC]
{
    extern __shared__ char smem[];
    __bf16* sD   = (__bf16*)(smem + SD_OFF);    // row-major [c][a], stride AA
    __bf16* sCVT = (__bf16*)(smem + CVT_OFF);   // transposed CV: [b][a], stride CVST
    float*  sSig = (float*)(smem + SIG_OFF);
    float*  sWk  = (float*)(smem + SWK_OFF);
    float*  sDm  = (float*)(smem + SDM_OFF);

    const int k   = blockIdx.x;
    const int tid = threadIdx.x;
    const float Lam = LambdaP[0];

    // ---- fill phase -------------------------------------------------------
    if (tid < CPAD) sSig[tid] = 0.0f;
    sWk[tid] = W[k * AA + tid];                               // tid in [0,256)
    sDm[tid] = mean_t[k * AA + tid] - mean_s[k * AA + tid];
    __syncthreads();

    // D (bf16, padded rows -> 0): vectorized b128 load + b64 packed store
    for (int idx = tid; idx < CPAD * AA / 4; idx += 256) {
        const int c  = idx >> 6;            // 64 float4 per row
        const int a4 = (idx & 63) * 4;
        const float4 wk4 = *(const float4*)&sWk[a4];
        const float4 w4  = (c < CC) ? *(const float4*)&W[c * AA + a4] : wk4;
        v4bf d = { (__bf16)(w4.x - wk4.x), (__bf16)(w4.y - wk4.y),
                   (__bf16)(w4.z - wk4.z), (__bf16)(w4.w - wk4.w) };
        *(v4bf*)&sD[c * AA + a4] = d;
    }
    // CV^T (coalesced b128 global read, transposed b16 stores)
    const float* cvk = cov + (size_t)k * AA * AA;
    for (int idx = tid; idx < AA * AA / 4; idx += 256) {
        const int a  = idx >> 6;
        const int b4 = (idx & 63) * 4;
        const float4 cvv = *(const float4*)&cvk[a * AA + b4];
        sCVT[(b4 + 0) * CVST + a] = (__bf16)cvv.x;
        sCVT[(b4 + 1) * CVST + a] = (__bf16)cvv.y;
        sCVT[(b4 + 2) * CVST + a] = (__bf16)cvv.z;
        sCVT[(b4 + 3) * CVST + a] = (__bf16)cvv.w;
    }
    __syncthreads();

    // ---- WMMA phase: each wave owns ct in {wave, wave+8} ------------------
    const int wave = tid >> 5, lane = tid & 31;
    const int hi = lane >> 4, ln = lane & 15;

    for (int ct = wave; ct < 13; ct += 8) {
        // Cache all 8 A fragments (K=0..255) for row ct*16+ln in registers.
        // ISA 16-bit A layout: low lanes K a0+0..7 / a0+16..23; high lanes +8/+24.
        const __bf16* arow = sD + (ct * 16 + ln) * AA + hi * 8;
        v16bf afr[8];
#pragma unroll
        for (int i = 0; i < 8; ++i) {
            v8bf lo  = *(const v8bf*)(arow + 32 * i);
            v8bf hi8 = *(const v8bf*)(arow + 32 * i + 16);
            afr[i] = __builtin_shufflevector(lo, hi8,
                0,1,2,3,4,5,6,7,8,9,10,11,12,13,14,15);
        }

        float partial[8] = {0,0,0,0,0,0,0,0};

#pragma unroll 1
        for (int bt = 0; bt < 16; ++bt) {
            // Preload D[c][b] bits for the P.*D tail (issued first; retired
            // by the staged B-fragment waits -> wait-free tail).
            const int b = bt * 16 + ln;
            unsigned short draw[8];
#pragma unroll
            for (int v = 0; v < 8; ++v)
                draw[v] = *(const unsigned short*)
                          &sD[(ct * 16 + v + hi * 8) * AA + b];

            // B fragments: col n = ln; K matches A (lanes 0-15: a0+0..7 / a0+16..23,
            // lanes 16-31: a0+8..15 / a0+24..31).  B[k,n] = CV[a0+k][bt*16+n]
            const __bf16* brow = sCVT + (bt * 16 + ln) * CVST + hi * 8;
            v16bf bfr[8];
#pragma unroll
            for (int i = 0; i < 8; ++i) {
                v8bf lo = *(const v8bf*)(brow + 32 * i);
                v8bf h8 = *(const v8bf*)(brow + 32 * i + 16);
                bfr[i] = __builtin_shufflevector(lo, h8,
                    0,1,2,3,4,5,6,7,8,9,10,11,12,13,14,15);
            }

            v8f acc = {};
#pragma unroll
            for (int i = 0; i < 8; ++i)
                acc = __builtin_amdgcn_wmma_f32_16x16x32_bf16(
                    false, afr[i], false, bfr[i], (short)0, acc, false, false);

            // P tile lane layout (f32 C/D): m = v + 8*hi, n = ln.
#pragma unroll
            for (int v = 0; v < 8; ++v) {
                union { unsigned int u; float f; } cvt;
                cvt.u = (unsigned int)draw[v] << 16;     // bf16 -> f32
                partial[v] += acc[v] * cvt.f;
            }

            // Pin schedule: issue all 24 DS reads (8 u16 + 16 b128) before the
            // 8-WMMA chain so s_wait_dscnt counts are staged, not flushed to 0.
            __builtin_amdgcn_sched_group_barrier(0x100, 24, 0);  // DS reads
            __builtin_amdgcn_sched_group_barrier(0x008, 8, 0);   // WMMA group
        }
        // Fold: 16 lanes per c contribute their column partials (ds_add_f32).
#pragma unroll
        for (int v = 0; v < 8; ++v)
            atomicAdd(&sSig[ct * 16 + v + hi * 8], partial[v]);
    }

    // ---- shift (f32, cheap) ----------------------------------------------
    float shift = 0.0f;
    if (tid < CC) {
        const float* wr = W + tid * AA;
#pragma unroll 4
        for (int a = 0; a < AA; ++a) shift += (wr[a] - sWk[a]) * sDm[a];
    }
    __syncthreads();
    if (tid < CC)
        addOut[k * CC + tid] = 0.5f * Lam * sSig[tid] + Lam * shift;
}

// ---------------------------------------------------------------------------
// Block reduction helpers (wave32; 8 waves / 256 threads)
// ---------------------------------------------------------------------------
__device__ __forceinline__ float blockMax(float v, float* s) {
#pragma unroll
    for (int m = 16; m > 0; m >>= 1) v = fmaxf(v, __shfl_xor(v, m, 32));
    __syncthreads();
    if ((threadIdx.x & 31) == 0) s[threadIdx.x >> 5] = v;
    __syncthreads();
    float r = s[0];
#pragma unroll
    for (int i = 1; i < 8; ++i) r = fmaxf(r, s[i]);
    return r;
}

__device__ __forceinline__ float blockSum(float v, float* s) {
#pragma unroll
    for (int m = 16; m > 0; m >>= 1) v += __shfl_xor(v, m, 32);
    __syncthreads();
    if ((threadIdx.x & 31) == 0) s[threadIdx.x >> 5] = v;
    __syncthreads();
    float r = s[0];
#pragma unroll
    for (int i = 1; i < 8; ++i) r += s[i];
    return r;
}

// ---------------------------------------------------------------------------
// Kernel Z / H: zero accumulators, label histogram (counts as f32)
// ---------------------------------------------------------------------------
__global__ void rscm_zero_kernel(float* out, float* counts) {
    int i = threadIdx.x;
    if (i == 0) out[0] = 0.0f;
    if (i < CC) counts[i] = 0.0f;
}

__global__ void rscm_hist_kernel(const int* __restrict__ labels, float* counts) {
    int i = blockIdx.x * blockDim.x + threadIdx.x;
    if (i < NN) atomicAdd(&counts[labels[i]], 1.0f);
}

// ---------------------------------------------------------------------------
// Kernel 2: seesaw loss.  One block per sample n; thread c handles class c.
// ---------------------------------------------------------------------------
__global__ __launch_bounds__(256) void rscm_loss_kernel(
    const float* __restrict__ y_s,     // [NN,CC]
    const int*   __restrict__ labels,  // [NN]
    const float* __restrict__ add,     // [CC,CC]  (per-label logit addend)
    const float* __restrict__ counts,  // [CC]
    float* __restrict__ out)           // [1]
{
    __shared__ float sred[8];
    __shared__ float sSelf;

    const int n = blockIdx.x, tid = threadIdx.x;
    const int t = labels[n];
    const bool active = tid < CC;

    float aug = active ? (y_s[n * CC + tid] + add[t * CC + tid]) : -1e30f;

    // softmax(aug) -> scores
    float m1 = blockMax(aug, sred);
    float e  = active ? __expf(aug - m1) : 0.0f;
    float Z  = blockSum(e, sred);
    float score = e / Z;
    if (tid == t) sSelf = score;
    __syncthreads();
    float self = sSelf;

    // mitigation (class-count ratio ^ P) * compensation (score ratio ^ Q=2)
    float acct = fmaxf(counts[t], 1.0f);
    float accc = active ? fmaxf(counts[tid], 1.0f) : 1.0f;
    float r    = accc / acct;
    float mit  = (r < 1.0f) ? __powf(r, 0.8f) : 1.0f;
    float ratio = score / fmaxf(self, 0.01f);
    float comp  = (ratio > 1.0f) ? ratio * ratio : 1.0f;
    float seesaw = mit * comp;

    float logit = active ? (aug + ((tid != t) ? __logf(seesaw) : 0.0f)) : -1e30f;

    // log_softmax -> nll at target
    float m2 = blockMax(logit, sred);
    float e2 = active ? __expf(logit - m2) : 0.0f;
    float Z2 = blockSum(e2, sred);
    if (tid == t) {
        float logp = (logit - m2) - __logf(Z2);
        atomicAdd(out, -logp * (1.0f / (float)NN));
    }
}

// ---------------------------------------------------------------------------
// Launch
// ---------------------------------------------------------------------------
extern "C" void kernel_launch(void* const* d_in, const int* in_sizes, int n_in,
                              void* d_out, int out_size, void* d_ws, size_t ws_size,
                              hipStream_t stream) {
    const float* W       = (const float*)d_in[0];  // fc_weight [CC,AA]
    // d_in[1] features_source: unused numerically (matches reference)
    const float* y_s     = (const float*)d_in[2];  // [NN,CC]
    const int*   labels  = (const int*)  d_in[3];  // [NN]
    const float* Lambda  = (const float*)d_in[4];  // [1]
    const float* mean_s  = (const float*)d_in[5];  // [CC,AA]
    const float* mean_t  = (const float*)d_in[6];  // [CC,AA]
    const float* cov     = (const float*)d_in[7];  // [CC,AA,AA]
    float* out = (float*)d_out;

    // workspace: add table [CC*CC] f32, then counts [CC] f32
    float* add    = (float*)d_ws;
    float* counts = add + CC * CC;

    rscm_zero_kernel<<<1, 256, 0, stream>>>(out, counts);
    rscm_hist_kernel<<<(NN + 255) / 256, 256, 0, stream>>>(labels, counts);
    rscm_perclass_kernel<<<CC, 256, LDS_BYTES, stream>>>(
        W, mean_s, mean_t, cov, Lambda, add);
    rscm_loss_kernel<<<NN, 256, 0, stream>>>(y_s, labels, add, counts, out);
}
